// MambaIT_82059645157380
// MI455X (gfx1250) — matrix operand order repaired and checked by reading, not compiled, not run on hardware
//
#include <hip/hip_runtime.h>
#include <hip/hip_bf16.h>

// ---------------------------------------------------------------------------
// Mamba stack for MI455X (gfx1250): bf16 WMMA projections with async-LDS
// double-buffered B staging, register-pipelined A, 32x64 per-wave tiles,
// + register-resident selective scan.
// ---------------------------------------------------------------------------

typedef __bf16 bf16;
typedef __attribute__((ext_vector_type(16))) __bf16 v16bf;
typedef __attribute__((ext_vector_type(8)))  __bf16 v8bf;
typedef __attribute__((ext_vector_type(8)))  float  v8f;
typedef __attribute__((ext_vector_type(4)))  int    v4i;

#define DM     768          // d_model
#define DI     1536         // d_inner
#define DS     16           // d_state
#define DTR    48           // dt_rank
#define BB     2            // batch
#define LCn    256
#define LXn    1024
#define LL     (LCn + LXn)  // 1280
#define NROW   (BB * LL)    // 2560 rows (M for all GEMMs)
#define XZW    (2 * DI)     // 3072
#define DBLW   80           // dt_rank + 2*d_state
#define DBLP   128          // padded N for x_proj gemm (full 64-wide tiles)
#define DTK    64           // dt_rank padded to 64 so K % 32 == 0
#define NDEPTH 4

// ---- CDNA5 async global->LDS path (guarded; falls back to ld+ds_store) ----
#if defined(__has_builtin)
#  if __has_builtin(__builtin_amdgcn_global_load_async_to_lds_b128) && \
      __has_builtin(__builtin_amdgcn_s_wait_asynccnt)
#    define USE_ASYNC_LDS 1
#  endif
#endif
#ifndef USE_ASYNC_LDS
#  define USE_ASYNC_LDS 0
#endif

#if USE_ASYNC_LDS
typedef __attribute__((address_space(1))) v4i* gv4ip;   // global v4i*
typedef __attribute__((address_space(3))) v4i* lv4ip;   // LDS v4i*
#endif

__device__ __forceinline__ float sigmoidf_(float x) { return 1.f / (1.f + __expf(-x)); }

// ---------------------------------------------------------------------------
// WMMA GEMM:  C[M,N] (f32) = A[M,K] (bf16 row-major) * W[N,K]^T (bf16 row-major)
// Block = 256 threads (8 waves). Block tile 256M x 64N; each wave owns 32Mx64N
// (2 A frags x 4 B frags = 8 WMMAs per K-step; each LDS B frag reused twice).
// B tile (64x32 bf16 = 4KB) staged in LDS, double buffered, async copies
// (1 chunk per thread per stage). A fragments register-pipelined one K-step
// ahead so their global loads retire under the WMMAs.
// Requires: M % 256 == 0, N % 64 == 0, K % 32 == 0.
// ---------------------------------------------------------------------------
__global__ void __launch_bounds__(256)
k_gemm_bf16(const bf16* __restrict__ A, const bf16* __restrict__ W,
            float* __restrict__ C, int M, int N, int K)
{
    __shared__ __attribute__((aligned(32))) bf16 sB[2][64 * 32];

    const int lane = threadIdx.x & 31;
    const int wv   = threadIdx.x >> 5;
    const int m0   = blockIdx.x * 256 + wv * 32;   // wave's 32-row strip
    const int nb   = blockIdx.y * 64;
    const int hl   = lane >> 4;     // half-wave id
    const int ml   = lane & 15;

    // A fragment (16x32 bf16): lanes 0-15 -> K 0..7 & 16..23, lanes 16-31 -> 8..15 & 24..31
    const bf16* aP0 = A + (size_t)(m0 + ml)      * K + hl * 8;
    const bf16* aP1 = A + (size_t)(m0 + 16 + ml) * K + hl * 8;

    auto loadA = [&](const bf16* p, int k) -> v16bf {
        v8bf lo = *(const v8bf*)(p + k);
        v8bf hi = *(const v8bf*)(p + k + 16);
        return __builtin_shufflevector(lo, hi,
                0,1,2,3,4,5,6,7,8,9,10,11,12,13,14,15);
    };
    auto loadB = [&](const bf16* p) -> v16bf {
        v8bf lo = *(const v8bf*)p;
        v8bf hi = *(const v8bf*)(p + 16);
        return __builtin_shufflevector(lo, hi,
                0,1,2,3,4,5,6,7,8,9,10,11,12,13,14,15);
    };

    // Stage one 64x32 B tile into sB[buf]: 256 chunks of 16B, 1 per thread.
    auto stage = [&](int buf, int k) {
        const int c     = threadIdx.x;              // chunk id 0..255
        const int n_loc = c >> 2;                   // 0..63
        const int koff  = (c & 3) * 8;              // element offset
        const bf16* g = W + (size_t)(nb + n_loc) * K + k + koff;
        bf16*       l = &sB[buf][n_loc * 32 + koff];
#if USE_ASYNC_LDS
        __builtin_amdgcn_global_load_async_to_lds_b128(
            (gv4ip)(void*)g, (lv4ip)(void*)l, 0, 0);
#else
        *(v8bf*)l = *(const v8bf*)g;                // global_load_b128 + ds_store_b128
#endif
    };

    v8f acc0[4] = {v8f{}, v8f{}, v8f{}, v8f{}};
    v8f acc1[4] = {v8f{}, v8f{}, v8f{}, v8f{}};

    v16bf a0 = loadA(aP0, 0);
    v16bf a1 = loadA(aP1, 0);
    stage(0, 0);

    for (int k = 0; k < K; k += 32) {
        const int  cur  = (k >> 5) & 1;
        const bool more = (k + 32 < K);
        if (more) {
            stage(1 - cur, k + 32);
#if USE_ASYNC_LDS
            __builtin_amdgcn_s_wait_asynccnt(1);   // current buffer's copy retired
#endif
        } else {
#if USE_ASYNC_LDS
            __builtin_amdgcn_s_wait_asynccnt(0);
#endif
        }
        __syncthreads();

        // pipeline A for the next K-step (retires under this step's WMMAs)
        v16bf na0 = a0, na1 = a1;
        if (more) {
            na0 = loadA(aP0, k + 32);
            na1 = loadA(aP1, k + 32);
            __builtin_prefetch(aP0 + k + 96, 0, 0);   // global_prefetch_b8
        }

        // B fragments from LDS: lane ml = column n, lanes 0-15 K 0..7 & 16..23, 16-31 K 8..15 & 24..31
        const bf16* sbase = &sB[cur][0];
        #pragma unroll
        for (int t = 0; t < 4; ++t) {
            const bf16* p = sbase + (t * 16 + ml) * 32 + hl * 8;
            v16bf b = loadB(p);
            acc0[t] = __builtin_amdgcn_wmma_f32_16x16x32_bf16(false, a0, false, b,
                                                              (short)0, acc0[t], false, false);
            acc1[t] = __builtin_amdgcn_wmma_f32_16x16x32_bf16(false, a1, false, b,
                                                              (short)0, acc1[t], false, false);
        }

        __syncthreads();   // all reads of sB[cur] done before it is re-staged
        a0 = na0;
        a1 = na1;
    }

    // C/D layout: VGPR r -> row base + 8*hl + r, column = nb + 16t + ml
    const int r0 = m0 + hl * 8;
    #pragma unroll
    for (int t = 0; t < 4; ++t) {
        const int c = nb + 16 * t + ml;
        #pragma unroll
        for (int r = 0; r < 8; ++r)
            C[(size_t)(r0 + r) * N + c] = acc0[t][r];
        #pragma unroll
        for (int r = 0; r < 8; ++r)
            C[(size_t)(r0 + 16 + r) * N + c] = acc1[t][r];
    }
}

// ---------------------------------------------------------------------------
// h = concat(context * mask, x)   (f32)
// ---------------------------------------------------------------------------
__global__ void __launch_bounds__(256)
k_build_h(float* __restrict__ h, const float* __restrict__ x,
          const float* __restrict__ ctx, const unsigned char* __restrict__ mask)
{
    size_t idx = (size_t)blockIdx.x * blockDim.x + threadIdx.x;
    if (idx >= (size_t)NROW * DM) return;
    int c   = (int)(idx % DM);
    size_t row = idx / DM;
    int l = (int)(row % LL);
    int b = (int)(row / LL);
    float v;
    if (l < LCn) {
        v = mask[b * LCn + l] ? ctx[((size_t)b * LCn + l) * DM + c] : 0.f;
    } else {
        v = x[((size_t)b * LXn + (l - LCn)) * DM + c];
    }
    h[idx] = v;
}

__global__ void __launch_bounds__(256)
k_cast_bf16(bf16* __restrict__ dst, const float* __restrict__ src, size_t n)
{
    size_t idx = (size_t)blockIdx.x * blockDim.x + threadIdx.x;
    if (idx < n) dst[idx] = (bf16)src[idx];
}

// dt_proj weights: [NDEPTH*DI, 48] -> [NDEPTH*DI, 64], zero-padded K
__global__ void __launch_bounds__(256)
k_pad_dtw(bf16* __restrict__ dst, const float* __restrict__ src)
{
    size_t idx = (size_t)blockIdx.x * blockDim.x + threadIdx.x;
    if (idx >= (size_t)NDEPTH * DI * DTK) return;
    int c = (int)(idx % DTK);
    size_t r = idx / DTK;
    dst[idx] = (c < DTR) ? (bf16)src[r * DTR + c] : (bf16)0.f;
}

// x_proj weights: [NDEPTH, 80, DI] -> [NDEPTH, 128, DI], zero-padded N rows
__global__ void __launch_bounds__(256)
k_pad_xpw(bf16* __restrict__ dst, const float* __restrict__ src)
{
    size_t idx = (size_t)blockIdx.x * blockDim.x + threadIdx.x;
    if (idx >= (size_t)NDEPTH * DBLP * DI) return;
    int kk = (int)(idx % DI);
    size_t r = idx / DI;
    int n   = (int)(r % DBLP);
    int lay = (int)(r / DBLP);
    dst[idx] = (n < DBLW) ? (bf16)src[((size_t)lay * DBLW + n) * DI + kk] : (bf16)0.f;
}

// ---------------------------------------------------------------------------
// Depthwise causal conv (width 4) + SiLU, reads xi half of xz. Emits f32 + bf16.
// ---------------------------------------------------------------------------
__global__ void __launch_bounds__(256)
k_conv_silu(const float* __restrict__ xz, const float* __restrict__ cw,
            const float* __restrict__ cb, float* __restrict__ u,
            bf16* __restrict__ ubf)
{
    size_t idx = (size_t)blockIdx.x * blockDim.x + threadIdx.x;
    if (idx >= (size_t)NROW * DI) return;
    int d = (int)(idx % DI);
    size_t row = idx / DI;
    int l = (int)(row % LL);
    size_t bBase = row - l;          // b*LL
    float acc = cb[d];
    #pragma unroll
    for (int k = 0; k < 4; ++k) {
        int j = l - 3 + k;
        if (j >= 0) acc += cw[d * 4 + k] * xz[(bBase + j) * XZW + d];
    }
    float s = acc * sigmoidf_(acc);
    u[idx]   = s;
    ubf[idx] = (bf16)s;
}

// dt activation (first 48 cols of dbl, stride DBLP) -> bf16 [NROW, 64], padded
__global__ void __launch_bounds__(256)
k_pack_dt(const float* __restrict__ dbl, bf16* __restrict__ dtbf)
{
    size_t idx = (size_t)blockIdx.x * blockDim.x + threadIdx.x;
    if (idx >= (size_t)NROW * DTK) return;
    int c = (int)(idx % DTK);
    size_t r = idx / DTK;
    dtbf[idx] = (c < DTR) ? (bf16)dbl[r * DBLP + c] : (bf16)0.f;
}

// dt = softplus(gemm + bias)  (in place on f32 buffer)
__global__ void __launch_bounds__(256)
k_softplus_bias(float* __restrict__ dtg, const float* __restrict__ dpb)
{
    size_t idx = (size_t)blockIdx.x * blockDim.x + threadIdx.x;
    if (idx >= (size_t)NROW * DI) return;
    int d = (int)(idx % DI);
    float v = dtg[idx] + dpb[d];
    dtg[idx] = (v > 20.f) ? v : log1pf(__expf(v));
}

// ---------------------------------------------------------------------------
// Selective scan: one thread per (b, d) channel; 16 states in registers;
// serial over L. Fused: + u*D, * z*sigmoid(z), bf16 output for out_proj.
// ---------------------------------------------------------------------------
__global__ void __launch_bounds__(256)
k_scan(const float* __restrict__ dtf, const float* __restrict__ dbl,
       const float* __restrict__ u, const float* __restrict__ xz,
       const float* __restrict__ alog, const float* __restrict__ Dp,
       bf16* __restrict__ ybf)
{
    int idx = blockIdx.x * blockDim.x + threadIdx.x;   // 0 .. BB*DI-1
    int d = idx % DI;
    int b = idx / DI;
    float Aa[DS], st[DS];
    #pragma unroll
    for (int n = 0; n < DS; ++n) {
        Aa[n] = -__expf(alog[d * DS + n]);
        st[n] = 0.f;
    }
    const float Dd = Dp[d];
    const size_t rowBase = (size_t)b * LL;
    for (int l = 0; l < LL; ++l) {
        const size_t r  = rowBase + l;
        const float dt  = dtf[r * DI + d];
        const float uu  = u[r * DI + d];
        const float* bc = dbl + r * DBLP;
        __builtin_prefetch(bc + DBLP, 0, 0);
        const float du  = dt * uu;
        float y = 0.f;
        #pragma unroll
        for (int n = 0; n < DS; ++n) {
            float s = __expf(dt * Aa[n]) * st[n] + du * bc[DTR + n];
            st[n] = s;
            y = fmaf(s, bc[DTR + DS + n], y);
        }
        y += uu * Dd;
        const float z = xz[r * XZW + DI + d];
        y *= z * sigmoidf_(z);
        ybf[r * DI + d] = (bf16)y;
    }
}

// ---------------------------------------------------------------------------
// h = rmsnorm(h + proj) * norm_w + 1e-6   (one block per row, LDS reduction)
// ---------------------------------------------------------------------------
__global__ void __launch_bounds__(256)
k_res_norm(float* __restrict__ h, const float* __restrict__ proj,
           const float* __restrict__ nw)
{
    __shared__ float red[256];
    const size_t row = blockIdx.x;
    float v[3];
    float ss = 0.f;
    #pragma unroll
    for (int j = 0; j < 3; ++j) {
        int c = threadIdx.x + j * 256;
        float t = h[row * DM + c] + proj[row * DM + c];
        v[j] = t;
        ss += t * t;
    }
    red[threadIdx.x] = ss;
    __syncthreads();
    #pragma unroll
    for (int s = 128; s > 0; s >>= 1) {
        if ((int)threadIdx.x < s) red[threadIdx.x] += red[threadIdx.x + s];
        __syncthreads();
    }
    const float scale = rsqrtf(red[0] / (float)DM + 1e-5f);
    #pragma unroll
    for (int j = 0; j < 3; ++j) {
        int c = threadIdx.x + j * 256;
        h[row * DM + c] = v[j] * scale * nw[c] + 1e-6f;
    }
}

// out = h[:, LC:, :]
__global__ void __launch_bounds__(256)
k_slice_out(float* __restrict__ out, const float* __restrict__ h)
{
    size_t idx = (size_t)blockIdx.x * blockDim.x + threadIdx.x;
    if (idx >= (size_t)BB * LXn * DM) return;
    int c = (int)(idx % DM);
    size_t rl = idx / DM;
    int l = (int)(rl % LXn);
    int b = (int)(rl / LXn);
    out[idx] = h[((size_t)b * LL + LCn + l) * DM + c];
}

// ---------------------------------------------------------------------------
#define CDIV(a, b) (((a) + (b) - 1) / (b))

extern "C" void kernel_launch(void* const* d_in, const int* in_sizes, int n_in,
                              void* d_out, int out_size, void* d_ws, size_t ws_size,
                              hipStream_t stream)
{
    (void)in_sizes; (void)n_in; (void)out_size; (void)ws_size;

    const float*          x     = (const float*)d_in[0];
    const float*          ctx   = (const float*)d_in[1];
    const unsigned char*  cmask = (const unsigned char*)d_in[2];
    const float*          ipw   = (const float*)d_in[3];
    const float*          cw    = (const float*)d_in[4];
    const float*          cb    = (const float*)d_in[5];
    const float*          xpw   = (const float*)d_in[6];
    const float*          dpw   = (const float*)d_in[7];
    const float*          dpb   = (const float*)d_in[8];
    const float*          alog  = (const float*)d_in[9];
    const float*          Dp    = (const float*)d_in[10];
    const float*          opw   = (const float*)d_in[11];
    const float*          nw    = (const float*)d_in[12];
    float*                out   = (float*)d_out;

    char* ws = (char*)d_ws;
    size_t off = 0;
    auto take = [&](size_t bytes) -> char* {
        off = (off + 255) & ~((size_t)255);
        char* p = ws + off;
        off += bytes;
        return p;
    };

    float* h    = (float*)take(sizeof(float) * (size_t)NROW * DM);
    bf16*  hbf  = (bf16*) take(2ull * NROW * DM);
    float* xz   = (float*)take(sizeof(float) * (size_t)NROW * XZW);
    float* u    = (float*)take(sizeof(float) * (size_t)NROW * DI);
    bf16*  ubf  = (bf16*) take(2ull * NROW * DI);
    float* dbl  = (float*)take(sizeof(float) * (size_t)NROW * DBLP);
    bf16*  dtbf = (bf16*) take(2ull * NROW * DTK);
    float* dtg  = (float*)take(sizeof(float) * (size_t)NROW * DI);
    bf16*  ybf  = (bf16*) take(2ull * NROW * DI);
    float* proj = (float*)take(sizeof(float) * (size_t)NROW * DM);
    bf16*  wip  = (bf16*) take(2ull * NDEPTH * XZW * DM);
    bf16*  wxp  = (bf16*) take(2ull * NDEPTH * DBLP * DI);
    bf16*  wdt  = (bf16*) take(2ull * NDEPTH * DI * DTK);
    bf16*  wop  = (bf16*) take(2ull * NDEPTH * DM * DI);

    // --- weight conversion to bf16 (every launch; deterministic) ---
    {
        size_t n;
        n = (size_t)NDEPTH * XZW * DM;
        k_cast_bf16<<<CDIV(n, 256), 256, 0, stream>>>(wip, ipw, n);
        n = (size_t)NDEPTH * DM * DI;
        k_cast_bf16<<<CDIV(n, 256), 256, 0, stream>>>(wop, opw, n);
        n = (size_t)NDEPTH * DBLP * DI;
        k_pad_xpw<<<CDIV(n, 256), 256, 0, stream>>>(wxp, xpw);
        n = (size_t)NDEPTH * DI * DTK;
        k_pad_dtw<<<CDIV(n, 256), 256, 0, stream>>>(wdt, dpw);
    }

    // --- h = concat(masked context, x) ---
    k_build_h<<<CDIV((size_t)NROW * DM, 256), 256, 0, stream>>>(h, x, ctx, cmask);

    for (int i = 0; i < NDEPTH; ++i) {
        // h -> bf16
        k_cast_bf16<<<CDIV((size_t)NROW * DM, 256), 256, 0, stream>>>(
            hbf, h, (size_t)NROW * DM);

        // xz = h @ in_proj_w^T   [2560 x 3072, K=768]
        k_gemm_bf16<<<dim3(NROW / 256, XZW / 64), 256, 0, stream>>>(
            hbf, wip + (size_t)i * XZW * DM, xz, NROW, XZW, DM);

        // u = silu(conv1d(xi))
        k_conv_silu<<<CDIV((size_t)NROW * DI, 256), 256, 0, stream>>>(
            xz, cw + (size_t)i * DI * 4, cb + (size_t)i * DI, u, ubf);

        // dbl = u @ x_proj_w^T   [2560 x 128(padded 80), K=1536]
        k_gemm_bf16<<<dim3(NROW / 256, DBLP / 64), 256, 0, stream>>>(
            ubf, wxp + (size_t)i * DBLP * DI, dbl, NROW, DBLP, DI);

        // dt raw -> bf16 padded
        k_pack_dt<<<CDIV((size_t)NROW * DTK, 256), 256, 0, stream>>>(dbl, dtbf);

        // dtg = dt @ dt_proj_w^T   [2560 x 1536, K=64]
        k_gemm_bf16<<<dim3(NROW / 256, DI / 64), 256, 0, stream>>>(
            dtbf, wdt + (size_t)i * DI * DTK, dtg, NROW, DI, DTK);

        // dtg = softplus(dtg + bias)
        k_softplus_bias<<<CDIV((size_t)NROW * DI, 256), 256, 0, stream>>>(
            dtg, dpb + (size_t)i * DI);

        // selective scan + gating -> ybf (bf16)
        k_scan<<<(BB * DI) / 256, 256, 0, stream>>>(
            dtg, dbl, u, xz, alog + (size_t)i * DI * DS, Dp + (size_t)i * DI, ybf);

        // proj = y @ out_proj_w^T   [2560 x 768, K=1536]
        k_gemm_bf16<<<dim3(NROW / 256, DM / 64), 256, 0, stream>>>(
            ybf, wop + (size_t)i * DM * DI, proj, NROW, DM, DI);

        // h = rmsnorm(h + proj) * nw + 1e-6
        k_res_norm<<<NROW, 256, 0, stream>>>(h, proj, nw);
    }

    k_slice_out<<<CDIV((size_t)BB * LXn * DM, 256), 256, 0, stream>>>(out, h);
}
